// R3DAttention_75385265979658
// MI455X (gfx1250) — matrix operand order repaired and checked by reading, not compiled, not run on hardware
//
#include <hip/hip_runtime.h>
#include <hip/hip_bf16.h>

// ---------------------------------------------------------------------------
// Shapes (fixed by the reference): b=2, c=1024, w=h=48 -> s=2304, 16 heads x 64
// ---------------------------------------------------------------------------
#define BATCH 2
#define CH    1024
#define WW    48
#define HH    48
#define SEQ   (WW * HH)        // 2304
#define BS    (BATCH * SEQ)    // 4608
#define C3    (3 * CH)         // 3072
#define NHEAD 16
#define HD    64
#define QT2   (SEQ / 32)       // 72 query tiles of 32

typedef __attribute__((ext_vector_type(16))) __bf16 v16bf;
typedef __attribute__((ext_vector_type(8)))  __bf16 v8bf;
typedef __attribute__((ext_vector_type(8)))  float  v8f;

union bf16x16 { v16bf v; v8bf h[2]; };

// B-operand load: lane holds one N-column, 16 contiguous K elements (32B).
__device__ inline v16bf load16(const __bf16* p) {
    bf16x16 u;
    u.h[0] = *(const v8bf*)(p);
    u.h[1] = *(const v8bf*)(p + 8);
    return u.v;
}

// A-operand load, contiguous K order (matches load16 B operands):
//   lanes 0-15  : row M=lane,    K = {0..15}
//   lanes 16-31 : row M=lane-16, K = {16..31}
__device__ inline v16bf load_a(const __bf16* row_k0, int half) {
    bf16x16 u;
    u.h[0] = *(const v8bf*)(row_k0 + half * 16);
    u.h[1] = *(const v8bf*)(row_k0 + half * 16 + 8);
    return u.v;
}

__device__ inline v8f bwmma(v16bf a, v16bf b, v8f c) {
    return __builtin_amdgcn_wmma_f32_16x16x32_bf16(
        false, a, false, b, (short)0, c, false, false);
}

// ---------------------------------------------------------------------------
// Kernel 1a: fp32 -> bf16 convert (weights)
// ---------------------------------------------------------------------------
__global__ void cvt_f32_bf16(const float* __restrict__ in,
                             __bf16* __restrict__ out, int n) {
    int i = blockIdx.x * blockDim.x + threadIdx.x;
    if (i < n) out[i] = (__bf16)in[i];
}

// ---------------------------------------------------------------------------
// Kernel 1b: x [b,c,w,h] -> xs_bf16 [b, s, c]
// ---------------------------------------------------------------------------
__global__ void permute_x(const float* __restrict__ x,
                          __bf16* __restrict__ xs) {
    int i = blockIdx.x * blockDim.x + threadIdx.x;      // < BS*CH, c fastest
    int c = i & (CH - 1);
    int t = (i >> 10) % SEQ;
    int b = i / (SEQ * CH);
    xs[i] = (__bf16)x[((size_t)b * CH + c) * SEQ + t];
}

// ---------------------------------------------------------------------------
// Kernel 2: QKV GEMM, register-blocked 32x64 tile per wave.
//   Per K-step: 2 A frags + 4 B frags (12 x b128) feed 8 wmmas.
//   Writes Q,K as [b,h,s,64]; V transposed as [b,h,64,s].
// ---------------------------------------------------------------------------
__global__ __launch_bounds__(128) void qkv_gemm(
    const __bf16* __restrict__ xs, const __bf16* __restrict__ wqkv,
    const float* __restrict__ bqkv,
    __bf16* __restrict__ qm, __bf16* __restrict__ km, __bf16* __restrict__ vt) {

    const int wave = threadIdx.x >> 5, lane = threadIdx.x & 31;
    const int half = lane >> 4, lm = lane & 15;
    const int m0 = blockIdx.x * 32;
    const int n0 = (blockIdx.y * 4 + wave) * 64;

    const __bf16* arow0 = xs + (size_t)(m0 + lm) * CH;
    const __bf16* arow1 = xs + (size_t)(m0 + 16 + lm) * CH;
    const __bf16* brow  = wqkv + (size_t)(n0 + lm) * CH + half * 16;

    v8f acc[2][4] = {};
#pragma unroll 4
    for (int k0 = 0; k0 < CH; k0 += 32) {
        v16bf a0 = load_a(arow0 + k0, half);
        v16bf a1 = load_a(arow1 + k0, half);
        v16bf b0 = load16(brow + k0);
        v16bf b1 = load16(brow + 16 * CH + k0);
        v16bf b2 = load16(brow + 32 * CH + k0);
        v16bf b3 = load16(brow + 48 * CH + k0);
        acc[0][0] = bwmma(a0, b0, acc[0][0]);
        acc[0][1] = bwmma(a0, b1, acc[0][1]);
        acc[0][2] = bwmma(a0, b2, acc[0][2]);
        acc[0][3] = bwmma(a0, b3, acc[0][3]);
        acc[1][0] = bwmma(a1, b0, acc[1][0]);
        acc[1][1] = bwmma(a1, b1, acc[1][1]);
        acc[1][2] = bwmma(a1, b2, acc[1][2]);
        acc[1][3] = bwmma(a1, b3, acc[1][3]);
    }

    // A 32-row tile never straddles a batch (SEQ % 32 == 0): hoist the divide.
    const int batch = m0 / SEQ;
    const int tok0  = m0 - batch * SEQ + 8 * half;   // fold in D-layout row base
    // n0 is a multiple of 64 -> one (three, head) per wave
    const int three = n0 >> 10;
    const int head  = (n0 & 1023) >> 6;
    const size_t hb = (size_t)batch * NHEAD + head;

    if (three < 2) {
        __bf16* dst = (three == 0 ? qm : km);
#pragma unroll
        for (int j = 0; j < 4; j++) {
            const int   d    = j * 16 + lm;
            const float bias = bqkv[n0 + d];
            __bf16* p = dst + (hb * SEQ + tok0) * HD + d;   // row stride HD
#pragma unroll
            for (int i = 0; i < 2; i++)
#pragma unroll
                for (int r = 0; r < 8; r++)
                    p[(size_t)(i * 16 + r) * HD] = (__bf16)(acc[i][j][r] + bias);
        }
    } else {
#pragma unroll
        for (int j = 0; j < 4; j++) {
            const int   d    = j * 16 + lm;
            const float bias = bqkv[n0 + d];
            __bf16* p = vt + (hb * HD + d) * SEQ + tok0;    // tok contiguous
#pragma unroll
            for (int i = 0; i < 2; i++)
#pragma unroll
                for (int r = 0; r < 8; r++)
                    p[i * 16 + r] = (__bf16)(acc[i][j][r] + bias);
        }
    }
}

// ---------------------------------------------------------------------------
// Kernel 3: flash attention, one wave per 32-query tile (2 subtiles of 16).
//   K/V fragments are reused by both query subtiles: 16 wmmas per 16 b128.
// ---------------------------------------------------------------------------
#define PSTRIDE 40   // padded bf16 row stride -> conflict-free b128 LDS reads

__global__ __launch_bounds__(128) void attention(
    const __bf16* __restrict__ qm, const __bf16* __restrict__ km,
    const __bf16* __restrict__ vt, __bf16* __restrict__ z) {

    __shared__ __bf16 Pl[4][32 * PSTRIDE];

    const int wave = threadIdx.x >> 5, lane = threadIdx.x & 31;
    const int half = lane >> 4, lm = lane & 15;
    const int wg = blockIdx.x * 4 + wave;
    const int qt = wg % QT2;
    const int head = (wg / QT2) & (NHEAD - 1);
    const int batch = wg / (QT2 * NHEAD);
    const int q0 = qt * 32;

    const size_t base = ((size_t)batch * NHEAD + head) * SEQ;
    const __bf16* vbase = vt + ((size_t)batch * NHEAD + head) * HD * SEQ;

    // Q fragments: 2 subtiles x 2 hd-chunks
    v16bf aq[2][2];
#pragma unroll
    for (int i = 0; i < 2; i++) {
        const __bf16* qrow = qm + (base + q0 + i * 16 + lm) * HD;
        aq[i][0] = load_a(qrow, half);
        aq[i][1] = load_a(qrow + 32, half);
    }

    v8f acc[2][4] = {};
    float mprev[2][8], lpart[2][8];
#pragma unroll
    for (int i = 0; i < 2; i++)
#pragma unroll
        for (int r = 0; r < 8; r++) { mprev[i][r] = -1e30f; lpart[i][r] = 0.0f; }

    __bf16* pw = &Pl[wave][0];
    const int nkb = (q0 + 31) / 32 + 1;          // causal block skip

    for (int kb = 0; kb < nkb; kb++) {
        const int ks = kb * 32;

        // ---- K fragments (shared by both query subtiles) ----
        v16bf bk[2][2];
#pragma unroll
        for (int t = 0; t < 2; t++) {
            const __bf16* kr = km + (base + ks + t * 16 + lm) * HD + half * 16;
            bk[t][0] = load16(kr);
            bk[t][1] = load16(kr + 32);
        }

        // ---- scores + online softmax per query subtile ----
#pragma unroll
        for (int i = 0; i < 2; i++) {
            v8f s0v = {}, s1v = {};
            s0v = bwmma(aq[i][0], bk[0][0], s0v);
            s0v = bwmma(aq[i][1], bk[0][1], s0v);
            s1v = bwmma(aq[i][0], bk[1][0], s1v);
            s1v = bwmma(aq[i][1], bk[1][1], s1v);
#pragma unroll
            for (int r = 0; r < 8; r++) {
                int qpos = q0 + i * 16 + r + 8 * half;
                int kp0 = ks + lm;
                float s0 = s0v[r] * 0.125f - ((kp0 > qpos) ? 10000.0f : 0.0f);
                float s1 = s1v[r] * 0.125f - ((kp0 + 16 > qpos) ? 10000.0f : 0.0f);
                float mb = fmaxf(s0, s1);
                mb = fmaxf(mb, __shfl_xor(mb, 1, 32));
                mb = fmaxf(mb, __shfl_xor(mb, 2, 32));
                mb = fmaxf(mb, __shfl_xor(mb, 4, 32));
                mb = fmaxf(mb, __shfl_xor(mb, 8, 32));
                float mn = fmaxf(mprev[i][r], mb);
                float scale = __expf(mprev[i][r] - mn);
                mprev[i][r] = mn;
                float p0 = __expf(s0 - mn);
                float p1 = __expf(s1 - mn);
                lpart[i][r] = lpart[i][r] * scale + p0 + p1;
                acc[i][0][r] *= scale; acc[i][1][r] *= scale;
                acc[i][2][r] *= scale; acc[i][3][r] *= scale;
                int M = i * 16 + r + 8 * half;
                pw[M * PSTRIDE + lm]      = (__bf16)p0;
                pw[M * PSTRIDE + 16 + lm] = (__bf16)p1;
            }
        }

        // ---- V fragments (shared by both query subtiles) ----
        v16bf bv[4];
#pragma unroll
        for (int j = 0; j < 4; j++)
            bv[j] = load16(vbase + (size_t)(j * 16 + lm) * SEQ + ks + half * 16);

        // ---- P @ V (intra-wave LDS ops are in-order; no barrier needed) ----
#pragma unroll
        for (int i = 0; i < 2; i++) {
            v16bf ap = load_a(pw + (i * 16 + lm) * PSTRIDE, half);
            acc[i][0] = bwmma(ap, bv[0], acc[i][0]);
            acc[i][1] = bwmma(ap, bv[1], acc[i][1]);
            acc[i][2] = bwmma(ap, bv[2], acc[i][2]);
            acc[i][3] = bwmma(ap, bv[3], acc[i][3]);
        }
    }

    // ---- finalize: reduce distributed row sums, normalize, store z bf16 ----
#pragma unroll
    for (int i = 0; i < 2; i++) {
#pragma unroll
        for (int r = 0; r < 8; r++) {
            float L = lpart[i][r];
            L += __shfl_xor(L, 1, 32);
            L += __shfl_xor(L, 2, 32);
            L += __shfl_xor(L, 4, 32);
            L += __shfl_xor(L, 8, 32);
            float inv = 1.0f / L;
            int tok = q0 + i * 16 + r + 8 * half;
            size_t zrow = ((size_t)batch * SEQ + tok) * CH + head * HD;
#pragma unroll
            for (int j = 0; j < 4; j++)
                z[zrow + j * 16 + lm] = (__bf16)(acc[i][j][r] * inv);
        }
    }
}

// ---------------------------------------------------------------------------
// Kernel 4: output projection, register-blocked 32x64 tile per wave.
//   out [b,c,w,h] fp32
// ---------------------------------------------------------------------------
__global__ __launch_bounds__(128) void out_proj(
    const __bf16* __restrict__ z, const __bf16* __restrict__ wo,
    const float* __restrict__ bo, float* __restrict__ out) {

    const int wave = threadIdx.x >> 5, lane = threadIdx.x & 31;
    const int half = lane >> 4, lm = lane & 15;
    const int m0 = blockIdx.x * 32;
    const int n0 = (blockIdx.y * 4 + wave) * 64;

    const __bf16* arow0 = z + (size_t)(m0 + lm) * CH;
    const __bf16* arow1 = z + (size_t)(m0 + 16 + lm) * CH;
    const __bf16* brow  = wo + (size_t)(n0 + lm) * CH + half * 16;

    v8f acc[2][4] = {};
#pragma unroll 4
    for (int k0 = 0; k0 < CH; k0 += 32) {
        v16bf a0 = load_a(arow0 + k0, half);
        v16bf a1 = load_a(arow1 + k0, half);
        v16bf b0 = load16(brow + k0);
        v16bf b1 = load16(brow + 16 * CH + k0);
        v16bf b2 = load16(brow + 32 * CH + k0);
        v16bf b3 = load16(brow + 48 * CH + k0);
        acc[0][0] = bwmma(a0, b0, acc[0][0]);
        acc[0][1] = bwmma(a0, b1, acc[0][1]);
        acc[0][2] = bwmma(a0, b2, acc[0][2]);
        acc[0][3] = bwmma(a0, b3, acc[0][3]);
        acc[1][0] = bwmma(a1, b0, acc[1][0]);
        acc[1][1] = bwmma(a1, b1, acc[1][1]);
        acc[1][2] = bwmma(a1, b2, acc[1][2]);
        acc[1][3] = bwmma(a1, b3, acc[1][3]);
    }

    // Tile never straddles a batch: hoist divide, use imm-offset store groups.
    const int batch = m0 / SEQ;
    const int tok0  = m0 - batch * SEQ + 8 * half;

#pragma unroll
    for (int j = 0; j < 4; j++) {
        const int   col  = n0 + j * 16 + lm;
        const float bias = bo[col];
        float* p = out + ((size_t)batch * CH + col) * SEQ + tok0;  // tok contig
#pragma unroll
        for (int i = 0; i < 2; i++)
#pragma unroll
            for (int r = 0; r < 8; r++)
                p[i * 16 + r] = acc[i][j][r] + bias;
    }
}

// ---------------------------------------------------------------------------
// Launcher
// ---------------------------------------------------------------------------
extern "C" void kernel_launch(void* const* d_in, const int* in_sizes, int n_in,
                              void* d_out, int out_size, void* d_ws, size_t ws_size,
                              hipStream_t stream) {
    const float* x    = (const float*)d_in[0];
    const float* Wqkv = (const float*)d_in[1];
    const float* bqkv = (const float*)d_in[2];
    const float* Wo   = (const float*)d_in[3];
    const float* bo   = (const float*)d_in[4];
    float* out = (float*)d_out;

    char* ws = (char*)d_ws;
    __bf16* xs    = (__bf16*)ws;  ws += (size_t)BS * CH * 2;
    __bf16* wqkvb = (__bf16*)ws;  ws += (size_t)C3 * CH * 2;
    __bf16* wob   = (__bf16*)ws;  ws += (size_t)CH * CH * 2;
    __bf16* qm    = (__bf16*)ws;  ws += (size_t)BATCH * NHEAD * SEQ * HD * 2;
    __bf16* km    = (__bf16*)ws;  ws += (size_t)BATCH * NHEAD * SEQ * HD * 2;
    __bf16* vt    = (__bf16*)ws;  ws += (size_t)BATCH * NHEAD * HD * SEQ * 2;
    __bf16* z     = (__bf16*)ws;  ws += (size_t)BS * CH * 2;

    cvt_f32_bf16<<<(C3 * CH + 255) / 256, 256, 0, stream>>>(Wqkv, wqkvb, C3 * CH);
    cvt_f32_bf16<<<(CH * CH + 255) / 256, 256, 0, stream>>>(Wo, wob, CH * CH);
    permute_x<<<(BS * CH + 255) / 256, 256, 0, stream>>>(x, xs);

    // QKV projection: 144 M-tiles x (12 blocks of 4 waves covering 3072 cols)
    qkv_gemm<<<dim3(BS / 32, C3 / 256), 128, 0, stream>>>(xs, wqkvb, bqkv, qm, km, vt);

    // flash attention: 2*16*72 query tiles, 4 waves/block
    attention<<<dim3(BATCH * NHEAD * QT2 / 4), 128, 0, stream>>>(qm, km, vt, z);

    // output projection: 144 x 4
    out_proj<<<dim3(BS / 32, CH / 256), 128, 0, stream>>>(z, wob, bo, out);
}
